// Mlp_ir_multi_80865644249614
// MI455X (gfx1250) — matrix operand order repaired and checked by reading, not compiled, not run on hardware
//
#include <hip/hip_runtime.h>
#include <hip/hip_bf16.h>

// ---------------------------------------------------------------------------
// Bidirectional flow-guided propagation (BasicVSR-style), CDNA5 / gfx1250.
// Per-pixel 288->96 GEMMs run on v_wmma_f32_16x16x32_bf16 (f32 accumulate).
// B=1, T=8, H=192, W=320, C=96.
// ---------------------------------------------------------------------------

typedef __attribute__((ext_vector_type(16))) __bf16 v16bf;
typedef __attribute__((ext_vector_type(8)))  float  v8f;

#define HH   192
#define WW   320
#define HWP  (HH*WW)          // 61440 pixels
#define CC   96
#define KD   288              // 3*C
#define KBN  9                // K blocks of 32
#define NBN  6                // N blocks of 16
#define FRAME ((size_t)HWP*CC)   // floats per frame

__device__ __forceinline__ void lds_fence() {
  // same-wave DS ops are in-order; this is a HW drain + compiler reorder fence
  asm volatile("s_wait_dscnt 0x0" ::: "memory");
}

union Frag  { v16bf v; uint4 u[2]; };
union Pack8 { __bf16 h[8]; uint4 u; };

__device__ __forceinline__ void stage8(__bf16* dst, float4 a, float4 b) {
  Pack8 t;
  t.h[0] = (__bf16)a.x; t.h[1] = (__bf16)a.y; t.h[2] = (__bf16)a.z; t.h[3] = (__bf16)a.w;
  t.h[4] = (__bf16)b.x; t.h[5] = (__bf16)b.y; t.h[6] = (__bf16)b.z; t.h[7] = (__bf16)b.w;
  *(uint4*)dst = t.u;   // ds_store_b128
}

__device__ __forceinline__ void zero_acc(v8f acc[NBN]) {
#pragma unroll
  for (int nb = 0; nb < NBN; ++nb)
#pragma unroll
    for (int r = 0; r < 8; ++r) acc[nb][r] = 0.0f;
}

// A panel (16 x 288 bf16) in LDS at st[m*KD + k]; weights pre-packed per-lane.
__device__ __forceinline__ void gemm288(const __bf16* st, const __bf16* wfrag,
                                        int lane, v8f acc[NBN]) {
  const int half = lane >> 4;
  const int mrow = lane & 15;
  const __bf16* arow = st + mrow*KD + half*8;   // A: K = kb*32 + (e/8)*16 + half*8 + e%8
  for (int kb = 0; kb < KBN; ++kb) {
    Frag a;
    a.u[0] = *(const uint4*)(arow + kb*32);        // K locals [half*8 .. +7]
    a.u[1] = *(const uint4*)(arow + kb*32 + 16);   // K locals [16+half*8 .. +7]
#pragma unroll
    for (int nb = 0; nb < NBN; ++nb) {
      Frag b;
      const __bf16* wp = wfrag + ((size_t)((kb*NBN + nb)*32 + lane))*16;
      b.u[0] = *(const uint4*)(wp);
      b.u[1] = *(const uint4*)(wp + 8);
      acc[nb] = __builtin_amdgcn_wmma_f32_16x16x32_bf16(
          false, a.v, false, b.v, (short)0, acc[nb], false, false);
    }
  }
}

// Pack W (288x96 row-major f32) into per-lane B fragments (bf16).
// B lane layout: n = nb*16 + (lane&15);  K = kb*32 + (e/8)*16 + (lane>>4)*8 + e%8.
__global__ void prep_weights(const float* w, __bf16* out, int total) {
  int idx = blockIdx.x*blockDim.x + threadIdx.x;
  if (idx >= total) return;
  int f    = idx >> 9;           // fragment (kb*6+nb)
  int lane = (idx >> 4) & 31;
  int e    = idx & 15;
  int kb = f / NBN, nb = f - kb*NBN;
  int half = lane >> 4, nl = lane & 15;
  int k = kb*32 + (e >> 3)*16 + half*8 + (e & 7);
  int n = nb*16 + nl;
  out[idx] = (__bf16)w[k*CC + n];
}

__global__ void copy4_kernel(float4* dst, const float4* src, int n) {
  int i = blockIdx.x*blockDim.x + threadIdx.x;
  int stride = gridDim.x*blockDim.x;
  for (; i < n; i += stride) dst[i] = src[i];
}

// One propagation step: (optional) warp + mean + mix-GEMM + GELU, then
// (optional) fusion GEMM, fully fused. 128 threads = 4 waves, 16 pixels/wave.
__global__ __launch_bounds__(128) void step_kernel(
    const float* x_t,          // x[:,i]            [HW][96]
    const float* warp_src,     // previous feat     [HW][96]
    const float* flow,         // [2][HW] (x-disp then y-disp)
    float*       sum_buf,      // running sum       [HW][96] (read + +=feat)
    float        inv_cnt,
    const __bf16* wmix, const float* bmix,
    float*       feat_out,     // mix result (f32)
    const float* fus_s0,       // out_back[i] source for fusion
    const __bf16* wfus, const float* bfus,
    float*       fus_out,      // final output frame (may alias fus_s0)
    int do_mix, int do_fus)
{
  __shared__ __align__(16) __bf16 stage[4][16*KD];   // 36,864 B
  const int lane  = threadIdx.x & 31;
  const int wave  = threadIdx.x >> 5;
  __bf16* st = stage[wave];
  const int pbase = (blockIdx.x*4 + wave) * 16;

  // staging role: 2 lanes per pixel, 48 channels each
  const int pl  = lane >> 1;
  const int c48 = (lane & 1) * 48;
  const int p   = pbase + pl;
  const int py  = p / WW;
  const int px  = p - py*WW;
  const float* xrow = x_t + (size_t)p*CC;
  __bf16* srow = st + pl*KD;

  const int half = lane >> 4;
  const int nl   = lane & 15;

  v8f acc[NBN];

  if (do_mix) {
    // ---- bilinear warp setup ----
    float fx = flow[p];
    float fy = flow[HWP + p];
    float gx = fminf(fmaxf((float)px + fx, 0.f), (float)(WW-1));
    float gy = fminf(fmaxf((float)py + fy, 0.f), (float)(HH-1));
    float fx0 = floorf(gx), fy0 = floorf(gy);
    int x0 = (int)fx0, y0 = (int)fy0;
    int x1 = min(x0+1, WW-1), y1 = min(y0+1, HH-1);
    float wx = gx - fx0, wy = gy - fy0;
    float w00 = (1.f-wy)*(1.f-wx), w01 = (1.f-wy)*wx;
    float w10 = wy*(1.f-wx),       w11 = wy*wx;
    const float* s00 = warp_src + (size_t)(y0*WW + x0)*CC;
    const float* s01 = warp_src + (size_t)(y0*WW + x1)*CC;
    const float* s10 = warp_src + (size_t)(y1*WW + x0)*CC;
    const float* s11 = warp_src + (size_t)(y1*WW + x1)*CC;
    const float* msrc = sum_buf + (size_t)p*CC;

    // ---- stage A panel: [x | guided | mean] as bf16 ----
#pragma unroll
    for (int cc = 0; cc < 48; cc += 8) {
      int c0 = c48 + cc;
      float4 xa = *(const float4*)(xrow + c0);
      float4 xb = *(const float4*)(xrow + c0 + 4);
      stage8(srow + c0, xa, xb);

      float4 a00 = *(const float4*)(s00 + c0), b00 = *(const float4*)(s00 + c0 + 4);
      float4 a01 = *(const float4*)(s01 + c0), b01 = *(const float4*)(s01 + c0 + 4);
      float4 a10 = *(const float4*)(s10 + c0), b10 = *(const float4*)(s10 + c0 + 4);
      float4 a11 = *(const float4*)(s11 + c0), b11 = *(const float4*)(s11 + c0 + 4);
      float4 ga, gb;
      ga.x = w00*a00.x + w01*a01.x + w10*a10.x + w11*a11.x;
      ga.y = w00*a00.y + w01*a01.y + w10*a10.y + w11*a11.y;
      ga.z = w00*a00.z + w01*a01.z + w10*a10.z + w11*a11.z;
      ga.w = w00*a00.w + w01*a01.w + w10*a10.w + w11*a11.w;
      gb.x = w00*b00.x + w01*b01.x + w10*b10.x + w11*b11.x;
      gb.y = w00*b00.y + w01*b01.y + w10*b10.y + w11*b11.y;
      gb.z = w00*b00.z + w01*b01.z + w10*b10.z + w11*b11.z;
      gb.w = w00*b00.w + w01*b01.w + w10*b10.w + w11*b11.w;
      stage8(srow + CC + c0, ga, gb);

      float4 ma = *(const float4*)(msrc + c0), mb = *(const float4*)(msrc + c0 + 4);
      ma.x *= inv_cnt; ma.y *= inv_cnt; ma.z *= inv_cnt; ma.w *= inv_cnt;
      mb.x *= inv_cnt; mb.y *= inv_cnt; mb.z *= inv_cnt; mb.w *= inv_cnt;
      stage8(srow + 2*CC + c0, ma, mb);
    }
    lds_fence();

    // ---- mix GEMM ----
    zero_acc(acc);
    gemm288(st, wmix, lane, acc);

    // ---- bias + exact GELU, write feat, update running sum, restage feat ----
#pragma unroll
    for (int nb = 0; nb < NBN; ++nb) {
      int c = nb*16 + nl;
      float bias = bmix[c];
#pragma unroll
      for (int r = 0; r < 8; ++r) {
        int m = r + 8*half;                      // C/D layout: M = r + 8*(lane/16)
        size_t idx = (size_t)(pbase + m)*CC + c;
        float z = acc[nb][r] + bias;
        float v = 0.5f*z*(1.f + erff(z*0.70710678118654752f));
        feat_out[idx] = v;
        sum_buf[idx]  = sum_buf[idx] + v;        // unique (pixel,chan) per lane
        if (do_fus) st[m*KD + CC + c] = (__bf16)v;   // fusion A section 1
      }
    }
  }

  if (do_fus) {
    lds_fence();
    const float* f0row = fus_s0 + (size_t)p*CC;
#pragma unroll
    for (int cc = 0; cc < 48; cc += 8) {
      int c0 = c48 + cc;
      float4 a = *(const float4*)(f0row + c0), b = *(const float4*)(f0row + c0 + 4);
      stage8(srow + c0, a, b);                        // section 0: out_back[i]
      float4 xa = *(const float4*)(xrow + c0), xb = *(const float4*)(xrow + c0 + 4);
      stage8(srow + 2*CC + c0, xa, xb);               // section 2: x[:,i]
      if (!do_mix) stage8(srow + CC + c0, xa, xb);    // step 0: feat == x0
    }
    lds_fence();

    zero_acc(acc);
    gemm288(st, wfus, lane, acc);

#pragma unroll
    for (int nb = 0; nb < NBN; ++nb) {
      int c = nb*16 + nl;
      float bias = bfus[c];
#pragma unroll
      for (int r = 0; r < 8; ++r) {
        int m = r + 8*half;
        size_t idx = (size_t)(pbase + m)*CC + c;
        fus_out[idx] = acc[nb][r] + bias;
      }
    }
  }
}

extern "C" void kernel_launch(void* const* d_in, const int* in_sizes, int n_in,
                              void* d_out, int out_size, void* d_ws, size_t ws_size,
                              hipStream_t stream) {
  const float* x   = (const float*)d_in[0];   // (1,8,192,320,96)
  const float* ff  = (const float*)d_in[1];   // (1,7,2,192,320)
  const float* fb  = (const float*)d_in[2];   // (1,7,2,192,320)
  const float* Wm  = (const float*)d_in[3];   // (288,96)
  const float* bm  = (const float*)d_in[4];   // (96,)
  const float* Wf  = (const float*)d_in[5];   // (288,96)
  const float* bfu = (const float*)d_in[6];   // (96,)
  float* out = (float*)d_out;                 // (1,8,192,320,96); doubles as out_back

  const size_t F = FRAME;                     // 5,898,240 floats / frame
  float* sumb   = (float*)d_ws;               // backward running sum
  float* fsum   = sumb + F;                   // forward running sum
  float* featf0 = fsum + F;                   // forward feat ping
  float* featf1 = featf0 + F;                 // forward feat pong
  __bf16* wmix  = (__bf16*)(featf1 + F);      // packed W_mix (54*512 bf16)
  __bf16* wfus  = wmix + 54*512;              // packed W_fus

  const int WTOT = 54*512;                    // 27648
  prep_weights<<<(WTOT+255)/256, 256, 0, stream>>>(Wm, wmix, WTOT);
  prep_weights<<<(WTOT+255)/256, 256, 0, stream>>>(Wf, wfus, WTOT);

  const int n4 = (int)(F/4);
  copy4_kernel<<<960,256,0,stream>>>((float4*)(out + 7*F), (const float4*)(x + 7*F), n4); // out_back[7]=x7
  copy4_kernel<<<960,256,0,stream>>>((float4*)sumb,        (const float4*)(x + 7*F), n4); // sum_b = x7
  copy4_kernel<<<960,256,0,stream>>>((float4*)featf0,      (const float4*)x,         n4); // feat_f0 = x0
  copy4_kernel<<<960,256,0,stream>>>((float4*)fsum,        (const float4*)x,         n4); // fsum = x0

  dim3 grid(HWP/64), block(128);   // 960 blocks, 4 waves x 16 pixels

  // ---- backward propagation: out_back[i] = feat_i, i = 6..0 ----
  for (int i = 6; i >= 0; --i) {
    step_kernel<<<grid, block, 0, stream>>>(
        x + (size_t)i*F, out + (size_t)(i+1)*F, fb + (size_t)i*2*HWP,
        sumb, 1.0f/(float)(7-i), wmix, bm,
        out + (size_t)i*F,
        nullptr, wfus, bfu, nullptr,
        /*do_mix=*/1, /*do_fus=*/0);
  }

  // ---- forward step 0: fusion only (feat = x0) ----
  step_kernel<<<grid, block, 0, stream>>>(
      x, x, fb, fsum, 1.0f, wmix, bm,
      featf1,
      out, wfus, bfu, out,
      /*do_mix=*/0, /*do_fus=*/1);

  // ---- forward steps 1..7: mix + fusion fused ----
  for (int i = 1; i <= 7; ++i) {
    float*       fo = (i & 1) ? featf1 : featf0;
    const float* wp = (i & 1) ? featf0 : featf1;
    step_kernel<<<grid, block, 0, stream>>>(
        x + (size_t)i*F, wp, ff + (size_t)(i-1)*2*HWP,
        fsum, 1.0f/(float)i, wmix, bm,
        fo,
        out + (size_t)i*F, wfus, bfu, out + (size_t)i*F,
        /*do_mix=*/1, /*do_fus=*/1);
  }
}
